// SpatialAttentionLayer_82927228551378
// MI455X (gfx1250) — matrix operand order repaired and checked by reading, not compiled, not run on hardware
//
#include <hip/hip_runtime.h>
#include <math.h>

// ---------------------------------------------------------------------------
// SpatialAttentionLayer fused kernel for gfx1250 (MI455X).
// B=8 T=128 K=64 d=256 H=8 dh=32 dff=1024 rb_h=32
// One workgroup (8 wave32s) per (b,t) tile; all GEMMs via v_wmma_f32_16x16x32_f16.
// ---------------------------------------------------------------------------

typedef __attribute__((ext_vector_type(16))) _Float16 v16h;
typedef __attribute__((ext_vector_type(8)))  _Float16 v8h;
typedef __attribute__((ext_vector_type(8)))  float    v8f;
typedef _Float16 half_t;

#define WMMA_F16(a, b, c) \
  __builtin_amdgcn_wmma_f32_16x16x32_f16(false, (a), false, (b), (short)0, (c), false, false)

// ---- packed-weight (d_ws) layout, offsets in halves -------------------------
// Global B-fragment layout (coalesced): frag = (kstep*ntiles + ntile)*512,
// within fragment: lane*16 + j  -> element (k = kstep*32 + (lane>>4)*16 + j, n = ntile*16 + (lane&15))
#define QKV_OFF 0        // 256 x 768  -> 8 ksteps * 48 ntiles
#define OUT_OFF 196608   // 256 x 256  -> 8 * 16
#define FF1_OFF 262144   // 256 x 1024 -> 8 * 64
#define FF2_OFF 524288   // 1024 x 256 -> 32 * 16
// total halves: 786432  (1.5 MB of d_ws)

// ---- dynamic LDS layout (bytes) ---------------------------------------------
#define SA_OFF     0         // 32KB  f16 A-frags (ksteps=8): Xn -> oh -> Yn
#define SQ_OFF     32768     // 32KB  f16 per-head q A-frags (h*2048 halves)
#define SK_OFF     65536     // 32KB  f16 per-head k^T B-frags
#define SV_OFF     98304     // 32KB  f16 per-head v B-frags
#define SBIAS_OFF  131072    // 16KB  f32 64x64 pairwise bias (M + Bt)
#define SP_OFF     147456    // 64KB  f16 per-head attn A-frags (h*4096 halves)
#define SH_OFF     32768     // 128KB f16 FF hidden A-frags (aliases SQ..SP; attn done)
#define SY_OFF     212992    // 64KB  f32 Y (64x256)
#define SW_OFF     278528    // 256B  W_t row
#define SBX_OFF    278784    // 768B  barX_t row
#define SRB_OFF    279552    // 400B  rb mlp params (w1[32],b1[32],w2[32],b2)
#define SMEM_BYTES 279952

__device__ __forceinline__ float gelu_exact(float x) {
  return 0.5f * x * (1.0f + erff(x * 0.7071067811865475f));
}

// ---- LDS fragment layout (bank-conflict-free) -------------------------------
// Within a 512-half fragment: element(lane, j) at ((j>>3)<<8) + (lane<<3) + (j&7)
// i.e. low 8 halves of all 32 lanes contiguous (16B/lane -> tiles all 64 banks),
// then the high 8 halves.  Loaded as two ds_load_b128 + shufflevector.
__device__ __forceinline__ int fidx(int ln, int j) {
  return ((j >> 3) << 8) + (ln << 3) + (j & 7);
}
// A-layout element (row m, col k) for a 64 x (32*ksteps) LDS region
__device__ __forceinline__ int aidx(int m, int k, int ksteps) {
  return (((m >> 4) * ksteps + (k >> 5)) << 9) +
         fidx((m & 15) + (((k >> 4) & 1) << 4), k & 15);
}
// Load one 16x32 f16 fragment (A or B operand) from LDS, frag = 512-half unit.
__device__ __forceinline__ v16h lds_frag(const char* smem, int region_byte_off,
                                         int frag, int lane) {
  const char* p = smem + region_byte_off + frag * 1024 + lane * 16;
  v8h lo = *(const v8h*)(p);
  v8h hi = *(const v8h*)(p + 512);
  return __builtin_shufflevector(lo, hi, 0, 1, 2, 3, 4, 5, 6, 7,
                                 8, 9, 10, 11, 12, 13, 14, 15);
}

// ============================================================================
// Kernel 0: pack fp32 weights into f16 WMMA B-fragment layout in d_ws.
// ============================================================================
__global__ void sal_pack_weights(const float* __restrict__ qkv_w,
                                 const float* __restrict__ out_w,
                                 const float* __restrict__ ff1_w,
                                 const float* __restrict__ ff2_w,
                                 half_t* __restrict__ wp) {
  int idx = blockIdx.x * 256 + threadIdx.x;
  if (idx < 196608) {                    // qkv_w : 256 x 768
    int k = idx / 768, n = idx - k * 768;
    wp[QKV_OFF + ((k >> 5) * 48 + (n >> 4)) * 512 + (((n & 15) | (((k >> 4) & 1) << 4)) << 4) + (k & 15)] =
        (half_t)qkv_w[idx];
    return;
  }
  idx -= 196608;
  if (idx < 65536) {                     // out_w : 256 x 256
    int k = idx >> 8, n = idx & 255;
    wp[OUT_OFF + ((k >> 5) * 16 + (n >> 4)) * 512 + (((n & 15) | (((k >> 4) & 1) << 4)) << 4) + (k & 15)] =
        (half_t)out_w[idx];
    return;
  }
  idx -= 65536;
  if (idx < 262144) {                    // ff1_w : 256 x 1024
    int k = idx >> 10, n = idx & 1023;
    wp[FF1_OFF + ((k >> 5) * 64 + (n >> 4)) * 512 + (((n & 15) | (((k >> 4) & 1) << 4)) << 4) + (k & 15)] =
        (half_t)ff1_w[idx];
    return;
  }
  idx -= 262144;
  if (idx < 262144) {                    // ff2_w : 1024 x 256
    int k = idx >> 8, n = idx & 255;
    wp[FF2_OFF + ((k >> 5) * 16 + (n >> 4)) * 512 + (((n & 15) | (((k >> 4) & 1) << 4)) << 4) + (k & 15)] =
        (half_t)ff2_w[idx];
  }
}

// ============================================================================
// Kernel 1: fused layer, one workgroup per (b,t) tile of 64 tokens.
// ============================================================================
__global__ void __launch_bounds__(256, 1)
sal_fused_kernel(const float* __restrict__ X_in, const float* __restrict__ barX,
                 const float* __restrict__ W_t,
                 const float* __restrict__ qkv_b, const float* __restrict__ out_b,
                 const float* __restrict__ ln1_g, const float* __restrict__ ln1_b,
                 const float* __restrict__ ln2_g, const float* __restrict__ ln2_b,
                 const float* __restrict__ ff1_b, const float* __restrict__ ff2_b,
                 const float* __restrict__ rb_w1, const float* __restrict__ rb_b1,
                 const float* __restrict__ rb_w2, const float* __restrict__ rb_b2,
                 const half_t* __restrict__ wp, float* __restrict__ out) {
  extern __shared__ __attribute__((aligned(128))) char smem[];
  const int tid  = threadIdx.x;
  const int wave = tid >> 5;
  const int lane = tid & 31;
  const int bt   = blockIdx.x;                 // 0..1023 = b*T + t
  const size_t xbase = (size_t)bt * 64 * 256;

  float* sW    = (float*)(smem + SW_OFF);
  float* sBX   = (float*)(smem + SBX_OFF);
  float* sRB   = (float*)(smem + SRB_OFF);
  float* sBias = (float*)(smem + SBIAS_OFF);
  float* sY    = (float*)(smem + SY_OFF);

  // ---- phase 0: stage per-tile smalls ----
  if (tid < 64)  sW[tid]  = W_t[(size_t)bt * 64 + tid];
  if (tid < 192) sBX[tid] = barX[(size_t)bt * 192 + tid];
  if (tid < 32) {
    sRB[tid]      = rb_w1[tid];
    sRB[32 + tid] = rb_b1[tid];
    sRB[64 + tid] = rb_w2[tid];
  }
  if (tid == 0) sRB[96] = rb_b2[0];
  __syncthreads();

  // ---- phase 1a: pairwise bias  M[i,j] + Bt[i,j]  (16 entries / thread) ----
  {
    const float rb2 = sRB[96];
    for (int e = 0; e < 16; ++e) {
      int idx = tid * 16 + e;
      int i = idx >> 6, j = idx & 63;
      float wi = sW[i], wj = sW[j];
      float gi = fminf(fmaxf(wi / (wi + 1e-3f), 0.f), 1.f);
      float gj = fminf(fmaxf(wj / (wj + 1e-3f), 0.f), 1.f);
      float Mij = logf(gi * gj + 1e-6f);
      float dx = sBX[i * 3 + 0] - sBX[j * 3 + 0];
      float dy = sBX[i * 3 + 1] - sBX[j * 3 + 1];
      float dz = sBX[i * 3 + 2] - sBX[j * 3 + 2];
      float dist = sqrtf(dx * dx + dy * dy + dz * dz);
      float acc = rb2;
#pragma unroll
      for (int h2 = 0; h2 < 32; ++h2)
        acc += gelu_exact(dist * sRB[h2] + sRB[32 + h2]) * sRB[64 + h2];
      sBias[idx] = Mij + acc;
    }
  }

  // ---- phase 1b: LN1 (4 adjacent lanes per row; shfl_xor reduction) ----
  const int row = tid >> 2, qq = tid & 3;
  const float* xr = X_in + xbase + (size_t)row * 256 + qq * 64;
  {
    float s = 0.f, ss = 0.f;
#pragma unroll
    for (int i = 0; i < 64; i += 4) {
      float4 v4 = *(const float4*)(xr + i);
      s  += v4.x + v4.y + v4.z + v4.w;
      ss += v4.x * v4.x + v4.y * v4.y + v4.z * v4.z + v4.w * v4.w;
    }
    s  += __shfl_xor(s, 1);  s  += __shfl_xor(s, 2);
    ss += __shfl_xor(ss, 1); ss += __shfl_xor(ss, 2);
    float mu   = s * (1.0f / 256.0f);
    float var  = ss * (1.0f / 256.0f) - mu * mu;
    float rsig = rsqrtf(var + 1e-5f);
    for (int i = 0; i < 64; i += 4) {
      float4 v4 = *(const float4*)(xr + i);
      float4 g4 = *(const float4*)(ln1_g + qq * 64 + i);
      float4 b4 = *(const float4*)(ln1_b + qq * 64 + i);
      int k = qq * 64 + i;
      *(half_t*)(smem + SA_OFF + aidx(row, k + 0, 8) * 2) = (half_t)((v4.x - mu) * rsig * g4.x + b4.x);
      *(half_t*)(smem + SA_OFF + aidx(row, k + 1, 8) * 2) = (half_t)((v4.y - mu) * rsig * g4.y + b4.y);
      *(half_t*)(smem + SA_OFF + aidx(row, k + 2, 8) * 2) = (half_t)((v4.z - mu) * rsig * g4.z + b4.z);
      *(half_t*)(smem + SA_OFF + aidx(row, k + 3, 8) * 2) = (half_t)((v4.w - mu) * rsig * g4.w + b4.w);
    }
  }
  __syncthreads();

  // ---- phase 2: QKV GEMM (64x256)@(256x768), split by section (q/k/v) ----
  // q section: ntile 0..15 -> per-head A-fragments, pre-scaled by 1/sqrt(dh)
  for (int tile = wave; tile < 64; tile += 8) {
    const int mtile = tile & 3;
    const int ntile = tile >> 2;                       // 0..15
    __builtin_prefetch(wp + QKV_OFF + ((ntile + 2 < 16 ? ntile + 2 : ntile)) * 512 + lane * 16, 0, 1);
    v8f acc = {};
#pragma unroll
    for (int ks = 0; ks < 8; ++ks) {
      v16h a  = lds_frag(smem, SA_OFF, mtile * 8 + ks, lane);
      v16h bm = *(const v16h*)(wp + QKV_OFF + (ks * 48 + ntile) * 512 + lane * 16);
      acc = WMMA_F16(a, bm, acc);
    }
    const int gn = ntile * 16 + (lane & 15);           // 0..255
    const float bv = qkv_b[gn];
    const int h = gn >> 5, c = gn & 31;
#pragma unroll
    for (int r = 0; r < 8; ++r) {
      int m = mtile * 16 + r + ((lane >> 4) << 3);
      float val = (acc[r] + bv) * 0.17677669529663687f;
      int ha = h * 2048 + (m >> 4) * 512 + fidx((m & 15) + (((c >> 4) & 1) << 4), c & 15);
      *(half_t*)(smem + SQ_OFF + ha * 2) = (half_t)val;
    }
  }
  // k section: ntile 16..31 -> per-head k^T B-fragments (per-element f16 stores)
  for (int tile = wave; tile < 64; tile += 8) {
    const int mtile = tile & 3;
    const int ntile = 16 + (tile >> 2);
    __builtin_prefetch(wp + QKV_OFF + ((ntile + 2 < 32 ? ntile + 2 : ntile)) * 512 + lane * 16, 0, 1);
    v8f acc = {};
#pragma unroll
    for (int ks = 0; ks < 8; ++ks) {
      v16h a  = lds_frag(smem, SA_OFF, mtile * 8 + ks, lane);
      v16h bm = *(const v16h*)(wp + QKV_OFF + (ks * 48 + ntile) * 512 + lane * 16);
      acc = WMMA_F16(a, bm, acc);
    }
    const int gnl = (ntile - 16) * 16 + (lane & 15);   // 0..255 within k
    const float bv = qkv_b[256 + gnl];
    const int h = gnl >> 5, c = gnl & 31;
#pragma unroll
    for (int r = 0; r < 8; ++r) {
      int m = mtile * 16 + r + ((lane >> 4) << 3);
      // fidx((m&15) + (c>>4)*16, c&15): lane = token, element = dh
      int ha0 = h * 2048 + mtile * 512 + fidx((m & 15) + ((c >> 4) << 4), c & 15);
      *(half_t*)(smem + SK_OFF + ha0 * 2) = (half_t)(acc[r] + bv);
    }
  }
  // v section: ntile 32..47 -> per-head v B-fragments (K=token, N=dh)
  for (int tile = wave; tile < 64; tile += 8) {
    const int mtile = tile & 3;
    const int ntile = 32 + (tile >> 2);
    __builtin_prefetch(wp + QKV_OFF + ((ntile + 2 < 48 ? ntile + 2 : ntile)) * 512 + lane * 16, 0, 1);
    v8f acc = {};
#pragma unroll
    for (int ks = 0; ks < 8; ++ks) {
      v16h a  = lds_frag(smem, SA_OFF, mtile * 8 + ks, lane);
      v16h bm = *(const v16h*)(wp + QKV_OFF + (ks * 48 + ntile) * 512 + lane * 16);
      acc = WMMA_F16(a, bm, acc);
    }
    const int gnl = (ntile - 32) * 16 + (lane & 15);
    const float bv = qkv_b[512 + gnl];
    const int h = gnl >> 5, c = gnl & 31;
#pragma unroll
    for (int r = 0; r < 8; ++r) {
      int m = mtile * 16 + r + ((lane >> 4) << 3);
      int ha = h * 2048 + ((m >> 5) * 2 + (c >> 4)) * 512 + fidx((c & 15) + (((m >> 4) & 1) << 4), m & 15);
      *(half_t*)(smem + SV_OFF + ha * 2) = (half_t)(acc[r] + bv);
    }
  }
  __syncthreads();

  // ---- phase 3: attention, one head per wave ----
  {
    const int h = wave;
    v8f S[16] = {};
#pragma unroll
    for (int mt = 0; mt < 4; ++mt)
#pragma unroll
      for (int nt = 0; nt < 4; ++nt) {
        v16h a  = lds_frag(smem, SQ_OFF, h * 4 + mt, lane);
        v16h bm = lds_frag(smem, SK_OFF, h * 4 + nt, lane);
        S[mt * 4 + nt] = WMMA_F16(a, bm, S[mt * 4 + nt]);
      }
    const int cb = lane & 15;
#pragma unroll
    for (int mt = 0; mt < 4; ++mt) {
#pragma unroll
      for (int r = 0; r < 8; ++r) {
        const int m = mt * 16 + r + ((lane >> 4) << 3);
        float v0 = S[mt * 4 + 0][r] + sBias[m * 64 +  0 + cb];
        float v1 = S[mt * 4 + 1][r] + sBias[m * 64 + 16 + cb];
        float v2 = S[mt * 4 + 2][r] + sBias[m * 64 + 32 + cb];
        float v3 = S[mt * 4 + 3][r] + sBias[m * 64 + 48 + cb];
        float mx = fmaxf(fmaxf(v0, v1), fmaxf(v2, v3));
        mx = fmaxf(mx, __shfl_xor(mx, 1));
        mx = fmaxf(mx, __shfl_xor(mx, 2));
        mx = fmaxf(mx, __shfl_xor(mx, 4));
        mx = fmaxf(mx, __shfl_xor(mx, 8));
        float e0 = __expf(v0 - mx), e1 = __expf(v1 - mx);
        float e2 = __expf(v2 - mx), e3 = __expf(v3 - mx);
        float sm = e0 + e1 + e2 + e3;
        sm += __shfl_xor(sm, 1);
        sm += __shfl_xor(sm, 2);
        sm += __shfl_xor(sm, 4);
        sm += __shfl_xor(sm, 8);
        float inv = 1.0f / sm;
#pragma unroll
        for (int nt = 0; nt < 4; ++nt) {
          float p = (nt == 0 ? e0 : nt == 1 ? e1 : nt == 2 ? e2 : e3) * inv;
          int l  = nt * 16 + cb;
          int ha = h * 4096 + (mt * 2 + (l >> 5)) * 512 +
                   fidx((m & 15) + (((l >> 4) & 1) << 4), l & 15);
          *(half_t*)(smem + SP_OFF + ha * 2) = (half_t)p;
        }
      }
    }
    // P @ V -> oh features [h*32, h*32+32) into sA A-fragments
#pragma unroll
    for (int mt = 0; mt < 4; ++mt)
#pragma unroll
      for (int nt = 0; nt < 2; ++nt) {
        v8f acc = {};
#pragma unroll
        for (int ks = 0; ks < 2; ++ks) {
          v16h a  = lds_frag(smem, SP_OFF, h * 8 + mt * 2 + ks, lane);
          v16h bm = lds_frag(smem, SV_OFF, h * 4 + ks * 2 + nt, lane);
          acc = WMMA_F16(a, bm, acc);
        }
#pragma unroll
        for (int r = 0; r < 8; ++r) {
          int m   = mt * 16 + r + ((lane >> 4) << 3);
          int n32 = nt * 16 + (lane & 15);
          int ha  = (mt * 8 + h) * 512 +
                    fidx((m & 15) + ((n32 >> 4) << 4), n32 & 15);
          *(half_t*)(smem + SA_OFF + ha * 2) = (half_t)acc[r];
        }
      }
  }
  __syncthreads();

  // ---- phase 4: out projection + residual -> sY ----
  for (int tile = wave; tile < 64; tile += 8) {
    const int mtile = tile & 3;
    const int ntile = tile >> 2;
    __builtin_prefetch(wp + OUT_OFF + ((ntile + 2 < 16 ? ntile + 2 : ntile)) * 512 + lane * 16, 0, 1);
    v8f acc = {};
#pragma unroll
    for (int ks = 0; ks < 8; ++ks) {
      v16h a  = lds_frag(smem, SA_OFF, mtile * 8 + ks, lane);
      v16h bm = *(const v16h*)(wp + OUT_OFF + (ks * 16 + ntile) * 512 + lane * 16);
      acc = WMMA_F16(a, bm, acc);
    }
    const int gn   = ntile * 16 + (lane & 15);
    const float ob = out_b[gn];
#pragma unroll
    for (int r = 0; r < 8; ++r) {
      int m = mtile * 16 + r + ((lane >> 4) << 3);
      sY[m * 256 + gn] = X_in[xbase + (size_t)m * 256 + gn] + acc[r] + ob;
    }
  }
  __syncthreads();

  // ---- phase 5: LN2 -> sA (Yn f16 A-fragments) ----
  {
    const float* yr = sY + row * 256 + qq * 64;
    float s = 0.f, ss = 0.f;
#pragma unroll
    for (int i = 0; i < 64; i += 4) {
      float4 v4 = *(const float4*)(yr + i);
      s  += v4.x + v4.y + v4.z + v4.w;
      ss += v4.x * v4.x + v4.y * v4.y + v4.z * v4.z + v4.w * v4.w;
    }
    s  += __shfl_xor(s, 1);  s  += __shfl_xor(s, 2);
    ss += __shfl_xor(ss, 1); ss += __shfl_xor(ss, 2);
    float mu   = s * (1.0f / 256.0f);
    float var  = ss * (1.0f / 256.0f) - mu * mu;
    float rsig = rsqrtf(var + 1e-5f);
    for (int i = 0; i < 64; i += 4) {
      float4 v4 = *(const float4*)(yr + i);
      float4 g4 = *(const float4*)(ln2_g + qq * 64 + i);
      float4 b4 = *(const float4*)(ln2_b + qq * 64 + i);
      int k = qq * 64 + i;
      *(half_t*)(smem + SA_OFF + aidx(row, k + 0, 8) * 2) = (half_t)((v4.x - mu) * rsig * g4.x + b4.x);
      *(half_t*)(smem + SA_OFF + aidx(row, k + 1, 8) * 2) = (half_t)((v4.y - mu) * rsig * g4.y + b4.y);
      *(half_t*)(smem + SA_OFF + aidx(row, k + 2, 8) * 2) = (half_t)((v4.z - mu) * rsig * g4.z + b4.z);
      *(half_t*)(smem + SA_OFF + aidx(row, k + 3, 8) * 2) = (half_t)((v4.w - mu) * rsig * g4.w + b4.w);
    }
  }
  __syncthreads();

  // ---- phase 6: FF1 + gelu -> sH (f16 A-fragments, ksteps=32) ----
  for (int tile = wave; tile < 256; tile += 8) {
    const int mtile = tile & 3;
    const int ntile = tile >> 2;           // 0..63
    __builtin_prefetch(wp + FF1_OFF + ((ntile + 2 < 64 ? ntile + 2 : ntile)) * 512 + lane * 16, 0, 1);
    v8f acc = {};
#pragma unroll
    for (int ks = 0; ks < 8; ++ks) {
      v16h a  = lds_frag(smem, SA_OFF, mtile * 8 + ks, lane);
      v16h bm = *(const v16h*)(wp + FF1_OFF + (ks * 64 + ntile) * 512 + lane * 16);
      acc = WMMA_F16(a, bm, acc);
    }
    const int gn   = ntile * 16 + (lane & 15);   // 0..1023
    const float fb = ff1_b[gn];
#pragma unroll
    for (int r = 0; r < 8; ++r) {
      int m = mtile * 16 + r + ((lane >> 4) << 3);
      float hv = gelu_exact(acc[r] + fb);
      int ha = (mtile * 32 + (gn >> 5)) * 512 +
               fidx((m & 15) + (((gn >> 4) & 1) << 4), gn & 15);
      *(half_t*)(smem + SH_OFF + ha * 2) = (half_t)hv;
    }
  }
  __syncthreads();

  // ---- phase 7: FF2 + residual -> out ----
  for (int tile = wave; tile < 64; tile += 8) {
    const int mtile = tile & 3;
    const int ntile = tile >> 2;
    __builtin_prefetch(wp + FF2_OFF + ((ntile + 2 < 16 ? ntile + 2 : ntile)) * 512 + lane * 16, 0, 1);
    v8f acc = {};
#pragma unroll
    for (int ks = 0; ks < 32; ++ks) {
      v16h a  = lds_frag(smem, SH_OFF, mtile * 32 + ks, lane);
      v16h bm = *(const v16h*)(wp + FF2_OFF + (ks * 16 + ntile) * 512 + lane * 16);
      acc = WMMA_F16(a, bm, acc);
    }
    const int gn    = ntile * 16 + (lane & 15);
    const float fb2 = ff2_b[gn];
#pragma unroll
    for (int r = 0; r < 8; ++r) {
      int m = mtile * 16 + r + ((lane >> 4) << 3);
      out[xbase + (size_t)m * 256 + gn] = sY[m * 256 + gn] + acc[r] + fb2;
    }
  }
}

// ============================================================================
extern "C" void kernel_launch(void* const* d_in, const int* in_sizes, int n_in,
                              void* d_out, int out_size, void* d_ws, size_t ws_size,
                              hipStream_t stream) {
  (void)in_sizes; (void)n_in; (void)out_size; (void)ws_size;
  const float* X_in  = (const float*)d_in[0];
  const float* barX  = (const float*)d_in[1];
  const float* W_t   = (const float*)d_in[2];
  const float* qkv_w = (const float*)d_in[3];
  const float* qkv_b = (const float*)d_in[4];
  const float* out_w = (const float*)d_in[5];
  const float* out_b = (const float*)d_in[6];
  const float* ln1_g = (const float*)d_in[7];
  const float* ln1_b = (const float*)d_in[8];
  const float* ln2_g = (const float*)d_in[9];
  const float* ln2_b = (const float*)d_in[10];
  const float* ff1_w = (const float*)d_in[11];
  const float* ff1_b = (const float*)d_in[12];
  const float* ff2_w = (const float*)d_in[13];
  const float* ff2_b = (const float*)d_in[14];
  const float* rb_w1 = (const float*)d_in[15];
  const float* rb_b1 = (const float*)d_in[16];
  const float* rb_w2 = (const float*)d_in[17];
  const float* rb_b2 = (const float*)d_in[18];
  half_t* wp  = (half_t*)d_ws;                // needs 1.5 MB of workspace
  float*  out = (float*)d_out;

  (void)hipFuncSetAttribute(reinterpret_cast<const void*>(sal_fused_kernel),
                            hipFuncAttributeMaxDynamicSharedMemorySize,
                            (int)SMEM_BYTES);

  sal_pack_weights<<<3072, 256, 0, stream>>>(qkv_w, out_w, ff1_w, ff2_w, wp);
  sal_fused_kernel<<<1024, 256, SMEM_BYTES, stream>>>(
      X_in, barX, W_t, qkv_b, out_b, ln1_g, ln1_b, ln2_g, ln2_b,
      ff1_b, ff2_b, rb_w1, rb_b1, rb_w2, rb_b2, wp, out);
}
